// Monotonic_attention_train_14035953123582
// MI455X (gfx1250) — matrix operand order repaired and checked by reading, not compiled, not run on hardware
//
#include <hip/hip_runtime.h>
#include <math.h>

// ---------------------------------------------------------------------------
// MI455X (gfx1250, wave32) monotonic-attention decoder training step.
// GEMMs: v_wmma_f32_16x16x32_bf16; fully-unrolled K loop (K, LDB compile-time)
// so all B-staging loads are saddr+imm form; A tile + B slab staged in LDS;
// guard-free inner loops (all N padded to multiples of 256).
// ---------------------------------------------------------------------------

typedef __attribute__((ext_vector_type(16))) __bf16 v16bf;
typedef __attribute__((ext_vector_type(8)))  float  v8f;

#define BLK 256

__device__ __forceinline__ __bf16 f2bf(float f) {
  unsigned u = __builtin_bit_cast(unsigned, f);
  unsigned r = u + 0x7FFFu + ((u >> 16) & 1u);   // round-to-nearest-even
  unsigned short h = (unsigned short)(r >> 16);
  __bf16 out;
  __builtin_memcpy(&out, &h, 2);
  return out;
}

__device__ __forceinline__ float sigmoidf_(float x) { return 1.f / (1.f + expf(-x)); }

// ------------------------------- casts -------------------------------------

__global__ void k_cast_bf16(const float* __restrict__ src, __bf16* __restrict__ dst, int n) {
  int i = blockIdx.x * blockDim.x + threadIdx.x;
  if (i < n) dst[i] = f2bf(src[i]);
}

__global__ void k_tanh_cast_bf16(const float* __restrict__ src, __bf16* __restrict__ dst, int n) {
  int i = blockIdx.x * blockDim.x + threadIdx.x;
  if (i < n) dst[i] = f2bf(tanhf(src[i]));
}

__global__ void k_cast_pad_bf16(const float* __restrict__ src, __bf16* __restrict__ dst,
                                int rows, int cols, int padcols) {
  int i = blockIdx.x * blockDim.x + threadIdx.x;
  int n = rows * padcols;
  if (i < n) {
    int r = i / padcols, c = i - r * padcols;
    dst[i] = f2bf(c < cols ? src[r * cols + c] : 0.f);
  }
}

// ----------------------------- small setup ---------------------------------

__global__ void k_vnorm(const float* __restrict__ v, const float* __restrict__ g,
                        float* __restrict__ out, int n) {
  __shared__ float red[BLK];
  float s = 0.f;
  for (int i = threadIdx.x; i < n; i += BLK) s += v[i] * v[i];
  red[threadIdx.x] = s;
  __syncthreads();
  for (int o = BLK / 2; o > 0; o >>= 1) {
    if (threadIdx.x < o) red[threadIdx.x] += red[threadIdx.x + o];
    __syncthreads();
  }
  if (threadIdx.x == 0) out[0] = g[0] / sqrtf(red[0]);
}

__global__ void k_init_state(float* __restrict__ aprev, float* __restrict__ s_state,
                             float* __restrict__ c_state, int BS, int S, int BH) {
  int i = blockIdx.x * blockDim.x + threadIdx.x;
  if (i < BS) aprev[i] = ((i % S) == 0) ? 1.f : 0.f;
  if (i < BH) { s_state[i] = 0.f; c_state[i] = 0.f; }
}

// --------------------------- WMMA bf16 GEMM --------------------------------
// C[M,N](f32) = A[M,K](bf16, lda==K) @ W[K,N](bf16, LDB)  [+C if ACC] [+bias]
// K, LDB compile-time -> fully unrolled slab loop, saddr+imm staging loads.
// Block: 256 threads (8 waves) = 16 rows x 256 cols, 2 N-tiles per wave.
// Host guarantees: N multiple of 256, M multiple of 16.

template <int K, int LDB, bool ACC, bool HASBIAS, bool BOUND>
__global__ __launch_bounds__(256)
void k_gemm_bf16(const __bf16* __restrict__ A,
                 const __bf16* __restrict__ Bm,
                 float* __restrict__ C, int ldc,
                 const float* __restrict__ bias, int nstore) {
  __shared__ __align__(16) __bf16 Atile[16 * K];
  __shared__ __align__(16) __bf16 Btile[256 * 32];   // 256 cols x 32 K, fragment layout

  const int tid  = threadIdx.x;
  const int wave = tid >> 5;
  const int lane = tid & 31;
  const int half = lane >> 4;            // 0: lanes 0-15, 1: lanes 16-31
  const int l16  = lane & 15;
  const int tileM   = blockIdx.y * 16;
  const int colbase = blockIdx.x * 256;
  const int lc0 = wave * 32;             // local col base of tile 0
  const int lc1 = lc0 + 16;
  const int n0  = colbase + lc0 + l16;
  const int n1  = n0 + 16;

  // ---- stage A tile: 16 rows x K bf16, uint4 copies, 32-bit indexing ----
  {
    const unsigned abase = (unsigned)tileM * K;
#pragma unroll
    for (int it = 0; it < (16 * K / 8) / BLK; it++) {
      unsigned idx = (unsigned)(it * BLK + tid) * 8;   // flat: rows contiguous (lda==K)
      *(uint4*)&Atile[idx] = *(const uint4*)&A[abase + idx];
    }
  }

  v8f acc0, acc1;
  if (ACC) {
#pragma unroll
    for (int r = 0; r < 8; r++) {
      acc0[r] = C[(size_t)(tileM + r + half * 8) * ldc + n0];
      acc1[r] = C[(size_t)(tileM + r + half * 8) * ldc + n1];
    }
  } else {
#pragma unroll
    for (int r = 0; r < 8; r++) { acc0[r] = 0.f; acc1[r] = 0.f; }
  }

  const unsigned short* __restrict__ Bs = (const unsigned short*)Bm;
  const int col = colbase + tid;         // this thread stages one column (always < N)

  // register-pipelined B staging; (k-const)*LDB folds into the imm offset
  unsigned wbuf[16];
#pragma unroll
  for (int j = 0; j < 16; j++) {
    unsigned lo = Bs[(2 * j) * LDB + col];
    unsigned hi = Bs[(2 * j + 1) * LDB + col];
    wbuf[j] = lo | (hi << 16);
  }

#pragma unroll
  for (int k0 = 0; k0 < K; k0 += 32) {
    __syncthreads();   // previous slab's fragment reads complete
    {
      uint4* dst = (uint4*)&Btile[tid * 32];
      dst[0] = make_uint4(wbuf[0],  wbuf[1],  wbuf[2],  wbuf[3]);
      dst[1] = make_uint4(wbuf[4],  wbuf[5],  wbuf[6],  wbuf[7]);
      dst[2] = make_uint4(wbuf[8],  wbuf[9],  wbuf[10], wbuf[11]);
      dst[3] = make_uint4(wbuf[12], wbuf[13], wbuf[14], wbuf[15]);
    }
    __syncthreads();   // slab ready

    if (k0 + 32 < K) {
      if (k0 + 64 < K)
        __builtin_prefetch(&Bs[(k0 + 64) * LDB + col], 0, 1);   // global_prefetch_b8
#pragma unroll
      for (int j = 0; j < 16; j++) {     // next slab's loads overlap with WMMA below
        unsigned lo = Bs[(k0 + 32 + 2 * j) * LDB + col];
        unsigned hi = Bs[(k0 + 32 + 2 * j + 1) * LDB + col];
        wbuf[j] = lo | (hi << 16);
      }
    }

    // A fragment (contiguous k = 16*half + s, matching B): one 32B LDS read
    v16bf af;
    __builtin_memcpy(&af, &Atile[l16 * K + k0 + half * 16], 32);

    // B fragments: one 32B contiguous LDS read per tile
    v16bf bf0, bf1;
    __builtin_memcpy(&bf0, &Btile[(lc0 + l16) * 32 + half * 16], 32);
    __builtin_memcpy(&bf1, &Btile[(lc1 + l16) * 32 + half * 16], 32);
    acc0 = __builtin_amdgcn_wmma_f32_16x16x32_bf16(false, af, false, bf0,
                                                   (short)0, acc0, false, false);
    acc1 = __builtin_amdgcn_wmma_f32_16x16x32_bf16(false, af, false, bf1,
                                                   (short)0, acc1, false, false);
  }

  // ---- epilogue ----
#pragma unroll
  for (int r = 0; r < 8; r++) {
    float v0 = acc0[r], v1 = acc1[r];
    if (HASBIAS) {
      if (!BOUND || n0 < nstore) v0 += bias[n0];
      if (!BOUND || n1 < nstore) v1 += bias[n1];
    }
    if (!BOUND || n0 < nstore) C[(size_t)(tileM + r + half * 8) * ldc + n0] = v0;
    if (!BOUND || n1 < nstore) C[(size_t)(tileM + r + half * 8) * ldc + n1] = v1;
  }
}

template <int K, int LDB, bool ACC, bool HASBIAS, bool BOUND = false>
static void launch_gemm(const __bf16* A, const __bf16* W, float* C, int ldc,
                        int M, int N, const float* bias, int nstore, hipStream_t s) {
  dim3 grid(N / 256, M / 16);
  k_gemm_bf16<K, LDB, ACC, HASBIAS, BOUND><<<grid, BLK, 0, s>>>(A, W, C, ldc, bias, nstore);
}

// ------------------------------ energies -----------------------------------
// One wave per (b,s) row; each lane reads a contiguous 64B (float4 x 4) chunk.

__global__ __launch_bounds__(256)
void k_energies(const float* __restrict__ enc_mono, const float* __restrict__ enc_chunk,
                const float* __restrict__ sm, const float* __restrict__ sc,
                const float* __restrict__ vmono, const float* __restrict__ vchunk,
                const float* __restrict__ vnorm_p, const float* __restrict__ rmono_p,
                const int* __restrict__ len, const float* __restrict__ noise,
                float* __restrict__ p_out, float* __restrict__ ech_out,
                int S, int D) {
  const int wave = threadIdx.x >> 5, lane = threadIdx.x & 31;
  const int row = blockIdx.x * 8 + wave;
  const int b = row / S, s = row - b * S;

  const float4* em4 = (const float4*)(enc_mono  + (size_t)row * D);
  const float4* ec4 = (const float4*)(enc_chunk + (size_t)row * D);
  const float4* sm4 = (const float4*)(sm + b * D);
  const float4* sc4 = (const float4*)(sc + b * D);
  const float4* vm4 = (const float4*)vmono;
  const float4* vc4 = (const float4*)vchunk;

  float am = 0.f, ac = 0.f;
#pragma unroll
  for (int j = 0; j < 4; j++) {          // D/4 = 128 float4; lane*4+j in [0,128)
    int idx = lane * 4 + j;
    float4 e = em4[idx], t = sm4[idx], v = vm4[idx];
    am += tanhf(e.x + t.x) * v.x + tanhf(e.y + t.y) * v.y +
          tanhf(e.z + t.z) * v.z + tanhf(e.w + t.w) * v.w;
    float4 e2 = ec4[idx], t2 = sc4[idx], v2 = vc4[idx];
    ac += tanhf(e2.x + t2.x) * v2.x + tanhf(e2.y + t2.y) * v2.y +
          tanhf(e2.z + t2.z) * v2.z + tanhf(e2.w + t2.w) * v2.w;
  }
#pragma unroll
  for (int o = 16; o > 0; o >>= 1) {
    am += __shfl_xor(am, o);
    ac += __shfl_xor(ac, o);
  }
  if (lane == 0) {
    bool masked = s >= len[b];
    if (masked) {
      p_out[row]   = 0.f;                // sigmoid(-inf + noise) == 0
      ech_out[row] = -__builtin_inff();
    } else {
      float e = vnorm_p[0] * am + rmono_p[0] + noise[b * S + s];
      p_out[row]   = sigmoidf_(e);
      ech_out[row] = ac;
    }
  }
}

// -------------------- monotonic recurrence + chunk attn --------------------

__global__ __launch_bounds__(256)
void k_recurrence(const float* __restrict__ p_g, const float* __restrict__ ech_g,
                  float* __restrict__ aprev_g, float* __restrict__ beta_g, int S) {
  const int b = blockIdx.x;
  const int t = threadIdx.x;
  const int PT = 4;                 // S / 256

  __shared__ float sh_p[1024], sh_ex[1024], sh_cpp[1024], sh_alpha[1024];
  __shared__ float sh_a[1024], sh_r[1024];
  __shared__ float part[BLK], red[BLK];

  for (int i = t; i < S; i += BLK) {
    sh_p[i]  = p_g[b * S + i];
    sh_ex[i] = ech_g[b * S + i];    // currently holds e_chunk
    sh_a[i]  = aprev_g[b * S + i];
  }
  __syncthreads();

  // --- max over e_chunk ---
  float mx = -__builtin_inff();
  for (int i = t; i < S; i += BLK) mx = fmaxf(mx, sh_ex[i]);
  red[t] = mx;
  __syncthreads();
  for (int o = BLK / 2; o > 0; o >>= 1) {
    if (t < o) red[t] = fmaxf(red[t], red[t + o]);
    __syncthreads();
  }
  const float m = red[0];
  __syncthreads();

  // --- ex = clip(exp(e - m), 1e-5, inf) ---
  for (int i = t; i < S; i += BLK) sh_ex[i] = fmaxf(expf(sh_ex[i] - m), 1e-5f);
  __syncthreads();

  // --- scan 1: cpp[i] = exp( cumsum_{j<i} log(clip(1-p[j],1e-8,1)) ) ---
  float loc[PT];
  float run = 0.f;
#pragma unroll
  for (int j = 0; j < PT; j++) {
    int i = t * PT + j;
    float lf = (i == 0) ? 0.f
                        : logf(fminf(fmaxf(1.f - sh_p[i - 1], 1e-8f), 1.f));
    run += lf;
    loc[j] = run;
  }
  part[t] = run;
  __syncthreads();
  for (int o = 1; o < BLK; o <<= 1) {
    float v = (t >= o) ? part[t - o] : 0.f;
    __syncthreads();
    part[t] += v;
    __syncthreads();
  }
  {
    float off = part[t] - run;      // exclusive prefix of this thread's chunk
#pragma unroll
    for (int j = 0; j < PT; j++) sh_cpp[t * PT + j] = expf(off + loc[j]);
  }
  __syncthreads();

  // --- scan 2: q = cpp * cumsum(a_prev / clip(cpp,1e-8,1)); alpha = clip(p*q) ---
  run = 0.f;
#pragma unroll
  for (int j = 0; j < PT; j++) {
    int i = t * PT + j;
    float u = sh_a[i] / fminf(fmaxf(sh_cpp[i], 1e-8f), 1.f);
    run += u;
    loc[j] = run;
  }
  part[t] = run;
  __syncthreads();
  for (int o = 1; o < BLK; o <<= 1) {
    float v = (t >= o) ? part[t - o] : 0.f;
    __syncthreads();
    part[t] += v;
    __syncthreads();
  }
  {
    float off = part[t] - run;
#pragma unroll
    for (int j = 0; j < PT; j++) {
      int i = t * PT + j;
      float q = sh_cpp[i] * (off + loc[j]);
      sh_alpha[i] = fminf(fmaxf(sh_p[i] * q, 1e-8f), 1.f);
    }
  }
  __syncthreads();

  // --- denom = clip(movsum(ex, back=3), 1e-10); r = alpha/denom ---
  for (int i = t; i < S; i += BLK) {
    float d = 0.f;
#pragma unroll
    for (int k = 0; k < 4; k++)
      if (i - k >= 0) d += sh_ex[i - k];
    d = fmaxf(d, 1e-10f);
    sh_r[i] = sh_alpha[i] / d;
  }
  __syncthreads();

  // --- beta = ex * movsum(r, fwd=3);  a_prev <- alpha ---
  for (int i = t; i < S; i += BLK) {
    float ms = 0.f;
#pragma unroll
    for (int k = 0; k < 4; k++)
      if (i + k < S) ms += sh_r[i + k];
    beta_g[b * S + i]  = sh_ex[i] * ms;
    aprev_g[b * S + i] = sh_alpha[i];
  }
}

// ------------------------------- context -----------------------------------
// ctx[b,d] = sum_s beta[b,s] * enc[b,s,d]

__global__ __launch_bounds__(256)
void k_ctx(const float* __restrict__ beta, const float* __restrict__ enc,
           float* __restrict__ ctx, int S, int D) {
  const int b = blockIdx.x;
  const int d = blockIdx.y * blockDim.x + threadIdx.x;
  const float* er = enc + (size_t)b * S * D + d;
  const float* br = beta + b * S;
  float acc = 0.f;
#pragma unroll 4
  for (int s = 0; s < S; s++) acc = fmaf(br[s], er[(size_t)s * D], acc);
  ctx[b * D + d] = acc;
}

// ------------------------------ LSTM gates ---------------------------------

__global__ __launch_bounds__(256)
void k_gates(const float* __restrict__ rec, const float* __restrict__ Lys,
             const int* __restrict__ target, int step, int nsteps,
             float* __restrict__ s_state, float* __restrict__ c_state, int H) {
  int tid = blockIdx.x * blockDim.x + threadIdx.x;   // B*H
  int b = tid / H, k = tid - b * H;
  int tgt = target[b * nsteps + step];
  const float* ly = Lys + (size_t)tgt * (4 * H);
  const float* r  = rec + (size_t)b * (4 * H);
  float i_ = r[k]         + ly[k];
  float f_ = r[H + k]     + ly[H + k];
  float g_ = r[2 * H + k] + ly[2 * H + k];
  float o_ = r[3 * H + k] + ly[3 * H + k];
  float c  = sigmoidf_(f_) * c_state[tid] + sigmoidf_(i_) * tanhf(g_);
  c_state[tid] = c;
  s_state[tid] = sigmoidf_(o_) * tanhf(c);
}

// ------------------------------- host side ---------------------------------

extern "C" void kernel_launch(void* const* d_in, const int* in_sizes, int n_in,
                              void* d_out, int out_size, void* d_ws, size_t ws_size,
                              hipStream_t stream) {
  const int Bn = 16, Sn = 1024, Hn = 256, Dn = 512, Cc = 1000, NS = 8;
  const int G4 = 4 * Hn;            // 1024
  const int CPAD = 1024;            // output GEMM padded to N multiple of 256

  const float* enc        = (const float*)d_in[0];
  const int*   target     = (const int*)d_in[1];
  const int*   len        = (const int*)d_in[2];
  const float* noise      = (const float*)d_in[3];
  const float* ws_mono_w  = (const float*)d_in[4];
  const float* ws_mono_b  = (const float*)d_in[5];
  const float* wh_mono    = (const float*)d_in[6];
  const float* v_mono     = (const float*)d_in[7];
  const float* g_mono     = (const float*)d_in[8];
  const float* r_mono     = (const float*)d_in[9];
  const float* ws_chunk_w = (const float*)d_in[10];
  const float* ws_chunk_b = (const float*)d_in[11];
  const float* wh_chunk   = (const float*)d_in[12];
  const float* v_chunk    = (const float*)d_in[13];
  const float* L_sy       = (const float*)d_in[14];
  const float* L_gy_w     = (const float*)d_in[15];
  const float* L_gy_b     = (const float*)d_in[16];
  const float* L_yy_w     = (const float*)d_in[17];
  const float* L_yy_b     = (const float*)d_in[18];
  const float* L_ys       = (const float*)d_in[19];
  const float* L_ss       = (const float*)d_in[20];
  const float* L_gs_w     = (const float*)d_in[21];
  const float* L_gs_b     = (const float*)d_in[22];
  float* out = (float*)d_out;

  // -------- workspace carve-out --------
  char* base = (char*)d_ws;
  size_t off = 0;
  auto alloc = [&](size_t bytes) -> char* {
    char* p = base + off;
    off += (bytes + 255) & ~(size_t)255;
    return p;
  };
  __bf16* enc_bf   = (__bf16*)alloc((size_t)Bn * Sn * Dn * 2);
  __bf16* whm_bf   = (__bf16*)alloc((size_t)Dn * Dn * 2);
  __bf16* whc_bf   = (__bf16*)alloc((size_t)Dn * Dn * 2);
  __bf16* wsm_bf   = (__bf16*)alloc((size_t)Hn * Dn * 2);
  __bf16* wsc_bf   = (__bf16*)alloc((size_t)Hn * Dn * 2);
  __bf16* lss_bf   = (__bf16*)alloc((size_t)Hn * G4 * 2);
  __bf16* lgs_bf   = (__bf16*)alloc((size_t)Dn * G4 * 2);
  __bf16* lgy_bf   = (__bf16*)alloc((size_t)Dn * Hn * 2);
  __bf16* lsy_bf   = (__bf16*)alloc((size_t)Hn * Hn * 2);
  __bf16* lyy_bf   = (__bf16*)alloc((size_t)Hn * CPAD * 2);
  float*  enc_mono = (float*)alloc((size_t)Bn * Sn * Dn * 4);
  float*  enc_chk  = (float*)alloc((size_t)Bn * Sn * Dn * 4);
  float*  p_buf    = (float*)alloc((size_t)Bn * Sn * 4);
  float*  ech_buf  = (float*)alloc((size_t)Bn * Sn * 4);
  float*  aprev    = (float*)alloc((size_t)Bn * Sn * 4);
  float*  beta     = (float*)alloc((size_t)Bn * Sn * 4);
  float*  sm       = (float*)alloc((size_t)Bn * Dn * 4);
  float*  sc       = (float*)alloc((size_t)Bn * Dn * 4);
  float*  s_state  = (float*)alloc((size_t)Bn * Hn * 4);
  float*  c_state  = (float*)alloc((size_t)Bn * Hn * 4);
  float*  ctx      = (float*)alloc((size_t)Bn * Dn * 4);
  float*  rec      = (float*)alloc((size_t)Bn * G4 * 4);
  float*  hpre     = (float*)alloc((size_t)Bn * Hn * 4);
  float*  vnorm    = (float*)alloc(256);
  __bf16* s_bf     = (__bf16*)alloc((size_t)Bn * Hn * 2);
  __bf16* snew_bf  = (__bf16*)alloc((size_t)Bn * Hn * 2);
  __bf16* ctx_bf   = (__bf16*)alloc((size_t)Bn * Dn * 2);
  __bf16* h_bf     = (__bf16*)alloc((size_t)Bn * Hn * 2);
  (void)ws_size; (void)in_sizes; (void)n_in; (void)out_size;

  auto cgrid = [](int n) { return (n + BLK - 1) / BLK; };

  // -------- one-time precompute --------
  k_cast_bf16<<<cgrid(Bn * Sn * Dn), BLK, 0, stream>>>(enc, enc_bf, Bn * Sn * Dn);
  k_cast_bf16<<<cgrid(Dn * Dn), BLK, 0, stream>>>(wh_mono, whm_bf, Dn * Dn);
  k_cast_bf16<<<cgrid(Dn * Dn), BLK, 0, stream>>>(wh_chunk, whc_bf, Dn * Dn);
  k_cast_bf16<<<cgrid(Hn * Dn), BLK, 0, stream>>>(ws_mono_w, wsm_bf, Hn * Dn);
  k_cast_bf16<<<cgrid(Hn * Dn), BLK, 0, stream>>>(ws_chunk_w, wsc_bf, Hn * Dn);
  k_cast_bf16<<<cgrid(Hn * G4), BLK, 0, stream>>>(L_ss, lss_bf, Hn * G4);
  k_cast_bf16<<<cgrid(Dn * G4), BLK, 0, stream>>>(L_gs_w, lgs_bf, Dn * G4);
  k_cast_bf16<<<cgrid(Dn * Hn), BLK, 0, stream>>>(L_gy_w, lgy_bf, Dn * Hn);
  k_cast_bf16<<<cgrid(Hn * Hn), BLK, 0, stream>>>(L_sy, lsy_bf, Hn * Hn);
  k_cast_pad_bf16<<<cgrid(Hn * CPAD), BLK, 0, stream>>>(L_yy_w, lyy_bf, Hn, Cc, CPAD);
  k_vnorm<<<1, BLK, 0, stream>>>(v_mono, g_mono, vnorm, Dn);
  k_init_state<<<cgrid(Bn * Sn), BLK, 0, stream>>>(aprev, s_state, c_state,
                                                   Bn * Sn, Sn, Bn * Hn);

  // enc_mono = enc @ Wh_mono ; enc_chunk = enc @ Wh_chunk   (M=16384,N=512,K=512)
  launch_gemm<512, 512, false, false>(enc_bf, whm_bf, enc_mono, Dn, Bn * Sn, Dn,
                                      nullptr, Dn, stream);
  launch_gemm<512, 512, false, false>(enc_bf, whc_bf, enc_chk, Dn, Bn * Sn, Dn,
                                      nullptr, Dn, stream);

  // -------- sequential decode steps --------
  for (int step = 0; step < NS; step++) {
    // s projections: sm/sc = s @ Ws_* + b   (M=16,N=512,K=256)
    k_cast_bf16<<<cgrid(Bn * Hn), BLK, 0, stream>>>(s_state, s_bf, Bn * Hn);
    launch_gemm<256, 512, false, true>(s_bf, wsm_bf, sm, Dn, Bn, Dn, ws_mono_b, Dn, stream);
    launch_gemm<256, 512, false, true>(s_bf, wsc_bf, sc, Dn, Bn, Dn, ws_chunk_b, Dn, stream);

    // energies + monotonic/chunk attention
    k_energies<<<(Bn * Sn) / 8, BLK, 0, stream>>>(
        enc_mono, enc_chk, sm, sc, v_mono, v_chunk, vnorm, r_mono, len,
        noise + (size_t)step * Bn * Sn, p_buf, ech_buf, Sn, Dn);
    k_recurrence<<<Bn, BLK, 0, stream>>>(p_buf, ech_buf, aprev, beta, Sn);
    k_ctx<<<dim3(Bn, Dn / BLK), BLK, 0, stream>>>(beta, enc, ctx, Sn, Dn);

    // LSTM recurrence: rec = s@L_ss + ctx@L_gs_w + L_gs_b   (M=16,N=1024)
    k_cast_bf16<<<cgrid(Bn * Dn), BLK, 0, stream>>>(ctx, ctx_bf, Bn * Dn);
    launch_gemm<256, 1024, false, false>(s_bf, lss_bf, rec, G4, Bn, G4, nullptr, G4, stream);
    launch_gemm<512, 1024, true, true>(ctx_bf, lgs_bf, rec, G4, Bn, G4, L_gs_b, G4, stream);
    k_gates<<<(Bn * Hn) / BLK, BLK, 0, stream>>>(rec, L_ys, target, step, NS,
                                                 s_state, c_state, Hn);

    // readout: y = tanh(ctx@L_gy_w + b + s_new@L_sy) @ L_yy_w + b
    k_cast_bf16<<<cgrid(Bn * Hn), BLK, 0, stream>>>(s_state, snew_bf, Bn * Hn);
    launch_gemm<512, 256, false, true>(ctx_bf, lgy_bf, hpre, Hn, Bn, Hn, L_gy_b, Hn, stream);
    launch_gemm<256, 256, true, false>(snew_bf, lsy_bf, hpre, Hn, Bn, Hn, nullptr, Hn, stream);
    k_tanh_cast_bf16<<<cgrid(Bn * Hn), BLK, 0, stream>>>(hpre, h_bf, Bn * Hn);
    // y[b, step, :] -> out + step*Cc (row stride NS*Cc), store only n < 1000
    launch_gemm<256, 1024, false, true, true>(h_bf, lyy_bf, out + (size_t)step * Cc,
                                              NS * Cc, Bn, CPAD, L_yy_b, Cc, stream);
  }
}
